// SuperpointNeuralOperator_3161095930016
// MI455X (gfx1250) — matrix operand order repaired and checked by reading, not compiled, not run on hardware
//
#include <hip/hip_runtime.h>
#include <hip/hip_bf16.h>
#include <math.h>

typedef __attribute__((ext_vector_type(16))) _Float16 v16h;
typedef __attribute__((ext_vector_type(8)))  _Float16 v8h;
typedef __attribute__((ext_vector_type(8)))  float    v8f;

#define HDIM 64
#define KNB  16
#define KPAD 160   // layer1 K padded 131 -> 160 (5 chunks of 32)
#define WPB  4     // waves (points) per 128-thread block

__device__ __forceinline__ float gelu_f(float x) {
    return 0.5f * x * (1.0f + erff(x * 0.70710678118654752f));
}
__device__ __forceinline__ float sigm_f(float x) {
    return 1.0f / (1.0f + expf(-x));
}

// Build a 16x32 f16 A-fragment for v_wmma_f32_16x16x32_f16 from an LDS row.
// Lane L (row = L&15, half = L>>4) holds K = {base+half*8 .. +7} and
// K = {base+16+half*8 .. +7} as a[0..7], a[8..15].
__device__ __forceinline__ v16h make_a(const _Float16* p) {
    v8h lo = *(const v8h*)(p);
    v8h hi = *(const v8h*)(p + 16);
    v16h a;
#pragma unroll
    for (int t = 0; t < 8; ++t) { a[t] = lo[t]; a[8 + t] = hi[t]; }
    return a;
}

// ---------------------------------------------------------------------------
// Weight pre-swizzle: f32 -> f16, K-padded, laid out so each lane reads its
// 16-element B-fragment (B[(t>>3)*16 + khalf*8 + (t&7)][n], n = ct*16 + (lane&15)) as 32
// contiguous bytes. K ordering is permuted to [v_i(64) | v_j(64) | rel(3) | 0]
// to make the LDS staging of kernel_input rows vector-aligned; the A-side
// staging uses the identical permutation, so the GEMM is unchanged.
// ---------------------------------------------------------------------------
__global__ void prep_weights(const float* __restrict__ gk_w1,
                             const float* __restrict__ gk_w2,
                             _Float16* __restrict__ w1sw,
                             _Float16* __restrict__ w2sw) {
    int tid = blockIdx.x * blockDim.x + threadIdx.x;
    if (tid < 5 * 4 * 32 * 16) {
        int t    = tid & 15;
        int lane = (tid >> 4) & 31;
        int ct   = (tid >> 9) & 3;
        int kc   = tid >> 11;
        int n    = ct * 16 + (lane & 15);
        int klds = kc * 32 + (t >> 3) * 16 + (lane >> 4) * 8 + (t & 7);     // 0..159
        float val = 0.0f;
        if (klds < 128)      val = gk_w1[(klds + 3) * HDIM + n];   // v_i, v_j
        else if (klds < 131) val = gk_w1[(klds - 128) * HDIM + n]; // rel_pos
        w1sw[tid] = (_Float16)val;
    }
    if (tid < 2 * 4 * 32 * 16) {
        int t    = tid & 15;
        int lane = (tid >> 4) & 31;
        int ct   = (tid >> 9) & 3;
        int kc   = tid >> 11;
        int n    = ct * 16 + (lane & 15);
        int k    = kc * 32 + (t >> 3) * 16 + (lane >> 4) * 8 + (t & 7);     // 0..63
        w2sw[tid] = (_Float16)gk_w2[k * HDIM + n];
    }
}

// ---------------------------------------------------------------------------
// v = [coords | feat] @ lift_w + lift_b   (f32 + f16 shadow)
// ---------------------------------------------------------------------------
__global__ void lift_kernel(const float* __restrict__ coords,
                            const float* __restrict__ feat,
                            const float* __restrict__ lift_w,
                            const float* __restrict__ lift_b,
                            float* __restrict__ v,
                            _Float16* __restrict__ vh, int N) {
    int t = blockIdx.x * blockDim.x + threadIdx.x;
    if (t >= N * HDIM) return;
    int i = t >> 6, c = t & 63;
    float a = lift_b[c];
#pragma unroll
    for (int d = 0; d < 3; ++d) a += coords[(size_t)i * 3 + d] * lift_w[d * HDIM + c];
#pragma unroll
    for (int d = 0; d < 6; ++d) a += feat[(size_t)i * 6 + d] * lift_w[(3 + d) * HDIM + c];
    v[t]  = a;
    vh[t] = (_Float16)a;
}

// ---------------------------------------------------------------------------
// Fused green-kernel MLP (WMMA) + integral + residual/W_w + ReLU + LayerNorm.
// MODE 0: message-passing update (write v_out/vh_out).
// MODE 1: final pass, write w_ij only.
// One wave per point: M-tile = its 16 neighbor rows.
// ---------------------------------------------------------------------------
template <int MODE>
__global__ __launch_bounds__(128)
void green_kernel(const float* __restrict__ coords,
                  const int* __restrict__ idx,
                  const float* __restrict__ v_in,
                  const _Float16* __restrict__ vh_in,
                  const _Float16* __restrict__ w1sw,
                  const _Float16* __restrict__ w2sw,
                  const float* __restrict__ gk_b1,
                  const float* __restrict__ gk_b2,
                  const float* __restrict__ gk_w3,
                  const float* __restrict__ gk_b3,
                  const float* __restrict__ W_w,
                  const float* __restrict__ ln_g,
                  const float* __restrict__ ln_b,
                  float* __restrict__ v_out,
                  _Float16* __restrict__ vh_out,
                  float* __restrict__ out_wij,
                  int N) {
    __shared__ __align__(64) _Float16 w1s[5 * 4 * 32 * 16];     // 20 KB
    __shared__ __align__(64) _Float16 w2s[2 * 4 * 32 * 16];     // 8 KB
    __shared__ float b1s[64], b2s[64], w3s[64];
    __shared__ __align__(64) _Float16 ki[WPB][16 * KPAD];       // 4 x 5 KB
    __shared__ __align__(64) _Float16 hb[WPB][16 * 64];         // 4 x 2 KB
    __shared__ float gv[WPB][16];
    __shared__ float vis[WPB][64];

    const int tid  = threadIdx.x;
    const int w    = tid >> 5;
    const int lane = tid & 31;
    const int half = lane >> 4;
    const int m    = lane & 15;

    // Stage swizzled weights + biases into LDS (block-cooperative).
    {
        const uint4* s1 = (const uint4*)w1sw; uint4* d1 = (uint4*)w1s;
        for (int q = tid; q < (5 * 4 * 32 * 16) / 8; q += 128) d1[q] = s1[q];
        const uint4* s2 = (const uint4*)w2sw; uint4* d2 = (uint4*)w2s;
        for (int q = tid; q < (2 * 4 * 32 * 16) / 8; q += 128) d2[q] = s2[q];
        if (tid < 64) { b1s[tid] = gk_b1[tid]; b2s[tid] = gk_b2[tid]; w3s[tid] = gk_w3[tid]; }
    }

    int i = blockIdx.x * WPB + w;
    const bool valid = (i < N);
    if (!valid) i = 0;                       // keep EXEC uniform for WMMA

    const int nb = idx[(size_t)i * KNB + m]; // neighbor of this lane's row

    // Stage kernel_input rows (f16): [0..63]=v_i  [64..127]=v_j  [128..130]=rel  pad=0
    {
        const _Float16* vhi = vh_in + (size_t)i * HDIM;
        const _Float16* vhj = vh_in + (size_t)nb * HDIM;
        _Float16* row = &ki[w][m * KPAD];
#pragma unroll
        for (int c = 0; c < 32; c += 8) {
            *(v8h*)&row[half * 32 + c]      = *(const v8h*)&vhi[half * 32 + c];
            *(v8h*)&row[64 + half * 32 + c] = *(const v8h*)&vhj[half * 32 + c];
        }
        if (half == 0) {
            float cx = coords[(size_t)i * 3 + 0];
            float cy = coords[(size_t)i * 3 + 1];
            float cz = coords[(size_t)i * 3 + 2];
            row[128] = (_Float16)(coords[(size_t)nb * 3 + 0] - cx);
            row[129] = (_Float16)(coords[(size_t)nb * 3 + 1] - cy);
            row[130] = (_Float16)(coords[(size_t)nb * 3 + 2] - cz);
            row[131] = (_Float16)0.0f;
        } else {
#pragma unroll
            for (int c = 132; c < 160; c += 2) *(unsigned int*)&row[c] = 0u;
        }
        vis[w][lane]      = v_in[(size_t)i * HDIM + lane];
        vis[w][lane + 32] = v_in[(size_t)i * HDIM + lane + 32];
    }
    __syncthreads();

    // ---- Layer 1: [16 x 160] @ [160 x 64], 5 K-chunks x 4 N-tiles = 20 WMMA
    v8f acc[4];
#pragma unroll
    for (int ct = 0; ct < 4; ++ct)
#pragma unroll
        for (int q = 0; q < 8; ++q) acc[ct][q] = 0.0f;

#pragma unroll
    for (int kc = 0; kc < 5; ++kc) {
        v16h a = make_a(&ki[w][m * KPAD + kc * 32 + half * 8]);
#pragma unroll
        for (int ct = 0; ct < 4; ++ct) {
            v16h b = *(const v16h*)&w1s[((kc * 4 + ct) * 32 + lane) * 16];
            acc[ct] = __builtin_amdgcn_wmma_f32_16x16x32_f16(
                false, a, false, b, (short)0, acc[ct], false, false);
        }
    }
    // bias + GELU -> hb (f16); C/D element (M=r+8*half, N=ct*16+m)
#pragma unroll
    for (int ct = 0; ct < 4; ++ct) {
        float bv = b1s[ct * 16 + m];
#pragma unroll
        for (int r = 0; r < 8; ++r) {
            float x = gelu_f(acc[ct][r] + bv);
            hb[w][(r + 8 * half) * 64 + ct * 16 + m] = (_Float16)x;
        }
    }
    __syncthreads();

    // ---- Layer 2: [16 x 64] @ [64 x 64], 2 K-chunks x 4 N-tiles = 8 WMMA
    v8f acc2[4];
#pragma unroll
    for (int ct = 0; ct < 4; ++ct)
#pragma unroll
        for (int q = 0; q < 8; ++q) acc2[ct][q] = 0.0f;

#pragma unroll
    for (int kc = 0; kc < 2; ++kc) {
        v16h a = make_a(&hb[w][m * 64 + kc * 32 + half * 8]);
#pragma unroll
        for (int ct = 0; ct < 4; ++ct) {
            v16h b = *(const v16h*)&w2s[((kc * 4 + ct) * 32 + lane) * 16];
            acc2[ct] = __builtin_amdgcn_wmma_f32_16x16x32_f16(
                false, a, false, b, (short)0, acc2[ct], false, false);
        }
    }

    // ---- Layer 3: GELU then 64 -> 1 dot + sigmoid (16-lane reductions)
    const float b3 = gk_b3[0];
    float pr[8];
#pragma unroll
    for (int r = 0; r < 8; ++r) {
        float s = 0.0f;
#pragma unroll
        for (int ct = 0; ct < 4; ++ct)
            s += gelu_f(acc2[ct][r] + b2s[ct * 16 + m]) * w3s[ct * 16 + m];
        s += __shfl_xor(s, 1, 16);
        s += __shfl_xor(s, 2, 16);
        s += __shfl_xor(s, 4, 16);
        s += __shfl_xor(s, 8, 16);
        pr[r] = s;
    }
    if (m == 0) {
#pragma unroll
        for (int r = 0; r < 8; ++r)
            gv[w][r + 8 * half] = sigm_f(pr[r] + b3);
    }
    __syncthreads();

    if (MODE == 1) {
        if (valid && lane < 16) out_wij[(size_t)i * KNB + lane] = gv[w][lane];
        return;
    }

    // ---- integral = mean_j G[j] * v_j  (channels c = lane, lane+32)
    float t0 = 0.0f, t1 = 0.0f;
#pragma unroll
    for (int j = 0; j < KNB; ++j) {
        float gj = gv[w][j];
        t0 += gj * (float)ki[w][j * KPAD + 64 + lane];
        t1 += gj * (float)ki[w][j * KPAD + 64 + lane + 32];
    }
    float u0 = t0 * (1.0f / 16.0f);
    float u1 = t1 * (1.0f / 16.0f);
    // + v_i @ W_w (f32, L2-resident)
#pragma unroll 8
    for (int d = 0; d < HDIM; ++d) {
        float vd = vis[w][d];
        u0 += vd * W_w[d * HDIM + lane];
        u1 += vd * W_w[d * HDIM + lane + 32];
    }
    u0 = fmaxf(u0, 0.0f);
    u1 = fmaxf(u1, 0.0f);

    // LayerNorm over 64 channels (wave reduction)
    float s = u0 + u1;
    s += __shfl_xor(s, 1, 32);  s += __shfl_xor(s, 2, 32);
    s += __shfl_xor(s, 4, 32);  s += __shfl_xor(s, 8, 32);
    s += __shfl_xor(s, 16, 32);
    float mu = s * (1.0f / 64.0f);
    float d0 = u0 - mu, d1 = u1 - mu;
    float q = d0 * d0 + d1 * d1;
    q += __shfl_xor(q, 1, 32);  q += __shfl_xor(q, 2, 32);
    q += __shfl_xor(q, 4, 32);  q += __shfl_xor(q, 8, 32);
    q += __shfl_xor(q, 16, 32);
    float inv = rsqrtf(q * (1.0f / 64.0f) + 1e-5f);
    float y0 = d0 * inv * ln_g[lane]      + ln_b[lane];
    float y1 = d1 * inv * ln_g[lane + 32] + ln_b[lane + 32];

    if (valid) {
        v_out[(size_t)i * HDIM + lane]       = y0;
        v_out[(size_t)i * HDIM + lane + 32]  = y1;
        vh_out[(size_t)i * HDIM + lane]      = (_Float16)y0;
        vh_out[(size_t)i * HDIM + lane + 32] = (_Float16)y1;
    }
}

// ---------------------------------------------------------------------------
// scores = sigmoid(gelu(v @ proj_w1 + b1) @ proj_w2 + b2); one wave per point.
// ---------------------------------------------------------------------------
__global__ __launch_bounds__(128)
void scores_kernel(const float* __restrict__ v,
                   const float* __restrict__ pw1, const float* __restrict__ pb1,
                   const float* __restrict__ pw2, const float* __restrict__ pb2,
                   float* __restrict__ out, int N) {
    int w = threadIdx.x >> 5, lane = threadIdx.x & 31;
    int i = blockIdx.x * WPB + w;
    if (i >= N) return;
    float a = pb1[lane];
    const float* vi = v + (size_t)i * HDIM;
#pragma unroll 8
    for (int d = 0; d < HDIM; ++d) a += vi[d] * pw1[d * 32 + lane];
    float s = gelu_f(a) * pw2[lane];
    s += __shfl_xor(s, 1, 32);  s += __shfl_xor(s, 2, 32);
    s += __shfl_xor(s, 4, 32);  s += __shfl_xor(s, 8, 32);
    s += __shfl_xor(s, 16, 32);
    if (lane == 0) out[i] = sigm_f(s + pb2[0]);
}

// idx (as float) and final v into d_out.
__global__ void finalize_kernel(const int* __restrict__ idx,
                                const float* __restrict__ vfin,
                                float* __restrict__ out_idx,
                                float* __restrict__ out_v,
                                int n_idx, int n_v) {
    int t = blockIdx.x * blockDim.x + threadIdx.x;
    if (t < n_idx) out_idx[t] = (float)idx[t];
    if (t < n_v)   out_v[t]   = vfin[t];
}

// ---------------------------------------------------------------------------
extern "C" void kernel_launch(void* const* d_in, const int* in_sizes, int n_in,
                              void* d_out, int out_size, void* d_ws, size_t ws_size,
                              hipStream_t stream) {
    const float* coords  = (const float*)d_in[0];
    const float* feat    = (const float*)d_in[1];
    const int*   idx     = (const int*)d_in[2];
    const float* lift_w  = (const float*)d_in[3];
    const float* lift_b  = (const float*)d_in[4];
    const float* gk_w1   = (const float*)d_in[5];
    const float* gk_b1   = (const float*)d_in[6];
    const float* gk_w2   = (const float*)d_in[7];
    const float* gk_b2   = (const float*)d_in[8];
    const float* gk_w3   = (const float*)d_in[9];
    const float* gk_b3   = (const float*)d_in[10];
    const float* W_w     = (const float*)d_in[11];
    const float* ln_g    = (const float*)d_in[12];
    const float* ln_b    = (const float*)d_in[13];
    const float* proj_w1 = (const float*)d_in[14];
    const float* proj_b1 = (const float*)d_in[15];
    const float* proj_w2 = (const float*)d_in[16];
    const float* proj_b2 = (const float*)d_in[17];

    const int N = in_sizes[0] / 3;   // 50000

    float* out        = (float*)d_out;
    float* out_scores = out;
    float* out_idx    = out + N;
    float* out_wij    = out + N + (size_t)N * KNB;
    float* out_v      = out + N + (size_t)2 * N * KNB;

    // workspace carve-out (256B-aligned)
    char* ws = (char*)d_ws;
    size_t off = 0;
    auto take = [&](size_t bytes) -> void* {
        void* p = ws + off;
        off += (bytes + 255) & ~(size_t)255;
        return p;
    };
    float*    vA   = (float*)take((size_t)N * HDIM * sizeof(float));
    float*    vB   = (float*)take((size_t)N * HDIM * sizeof(float));
    _Float16* vhA  = (_Float16*)take((size_t)N * HDIM * sizeof(_Float16));
    _Float16* vhB  = (_Float16*)take((size_t)N * HDIM * sizeof(_Float16));
    _Float16* w1sw = (_Float16*)take((size_t)5 * 4 * 32 * 16 * sizeof(_Float16));
    _Float16* w2sw = (_Float16*)take((size_t)2 * 4 * 32 * 16 * sizeof(_Float16));

    prep_weights<<<(5 * 4 * 32 * 16 + 255) / 256, 256, 0, stream>>>(gk_w1, gk_w2, w1sw, w2sw);
    lift_kernel<<<(N * HDIM + 255) / 256, 256, 0, stream>>>(coords, feat, lift_w, lift_b, vA, vhA, N);

    const int gb = (N + WPB - 1) / WPB;
    float* vin = vA;  float* vout = vB;
    _Float16* hin = vhA;  _Float16* hout = vhB;
    for (int t = 0; t < 3; ++t) {
        green_kernel<0><<<gb, 128, 0, stream>>>(
            coords, idx, vin, hin, w1sw, w2sw, gk_b1, gk_b2, gk_w3, gk_b3,
            W_w, ln_g + t * HDIM, ln_b + t * HDIM, vout, hout, nullptr, N);
        float* tv = vin; vin = vout; vout = tv;
        _Float16* th = hin; hin = hout; hout = th;
    }
    // final green pass for w_ij (uses final v)
    green_kernel<1><<<gb, 128, 0, stream>>>(
        coords, idx, vin, hin, w1sw, w2sw, gk_b1, gk_b2, gk_w3, gk_b3,
        W_w, ln_g, ln_b, vout, hout, out_wij, N);

    scores_kernel<<<gb, 128, 0, stream>>>(vin, proj_w1, proj_b1, proj_w2, proj_b2, out_scores, N);
    finalize_kernel<<<(N * HDIM + 255) / 256, 256, 0, stream>>>(idx, vin, out_idx, out_v,
                                                                N * KNB, N * HDIM);
}
